// TransformerControl_146028888616
// MI455X (gfx1250) — matrix operand order repaired and checked by reading, not compiled, not run on hardware
//
#include <hip/hip_runtime.h>
#include <hip/hip_bf16.h>
#include <math.h>

// ---------------------------------------------------------------------------
// MI455X (gfx1250) transformer stack: all GEMM-shaped math on
// v_wmma_f32_16x16x32_bf16 (wave32, fp32 accumulate), flash-attention with
// online softmax per wave (row-sums folded into an extra WMMA against a
// ones-matrix), weights converted fp32->bf16 while staging to LDS, and
// async global->LDS (ASYNCcnt) staging for the raw-copy paths.
// ---------------------------------------------------------------------------

typedef unsigned short u16;
typedef unsigned int   u32;
typedef unsigned long long u64;
typedef __attribute__((ext_vector_type(16))) __bf16 bf16x16;
typedef __attribute__((ext_vector_type(8)))  float  f32x8;

union Frag16 { bf16x16 v; uint4 q[2]; u16 s[16]; };

#define WMMA_BF16(A, Bm, C) \
  __builtin_amdgcn_wmma_f32_16x16x32_bf16(false, (A), false, (Bm), (short)0, (C), false, false)

__device__ __forceinline__ u16 f2bf(float f) {
  u32 u = __builtin_bit_cast(u32, f);
  u32 r = u + 0x7fffu + ((u >> 16) & 1u);   // round-to-nearest-even
  return (u16)(r >> 16);
}

// async global->LDS copy, 16B per lane (GLOBAL_LOAD_ASYNC_TO_LDS_B128).
// Generic LDS pointers carry the LDS byte offset in their low 32 bits.
__device__ __forceinline__ void async_ld_b128(void* lds_ptr, const void* gptr) {
  u32 l = (u32)(uintptr_t)lds_ptr;
  u64 g = (u64)(uintptr_t)gptr;
  asm volatile("global_load_async_to_lds_b128 %0, %1, off"
               :: "v"(l), "v"(g) : "memory");
}
__device__ __forceinline__ void wait_async0() {
  asm volatile("s_wait_asynccnt 0x0" ::: "memory");
}

// ------------------------------- LayerNorm ---------------------------------
__global__ __launch_bounds__(256) void ln_bf16_kernel(
    const float* __restrict__ h, const float* __restrict__ g,
    const float* __restrict__ b, u16* __restrict__ out) {
  __shared__ float s1[256];
  __shared__ float s2[256];
  const int row = blockIdx.x, tid = threadIdx.x;
  const float* hr = h + (size_t)row * 1024;
  float a0 = hr[tid], a1 = hr[tid + 256], a2 = hr[tid + 512], a3 = hr[tid + 768];
  s1[tid] = a0 + a1 + a2 + a3;
  s2[tid] = a0 * a0 + a1 * a1 + a2 * a2 + a3 * a3;
  __syncthreads();
  for (int o = 128; o > 0; o >>= 1) {
    if (tid < o) { s1[tid] += s1[tid + o]; s2[tid] += s2[tid + o]; }
    __syncthreads();
  }
  const float mu  = s1[0] * (1.0f / 1024.0f);
  const float var = s2[0] * (1.0f / 1024.0f) - mu * mu;
  const float rs  = rsqrtf(var + 1e-5f);
  u16* orow = out + (size_t)row * 1024;
  orow[tid      ] = f2bf((a0 - mu) * rs * g[tid      ] + b[tid      ]);
  orow[tid + 256] = f2bf((a1 - mu) * rs * g[tid + 256] + b[tid + 256]);
  orow[tid + 512] = f2bf((a2 - mu) * rs * g[tid + 512] + b[tid + 512]);
  orow[tid + 768] = f2bf((a3 - mu) * rs * g[tid + 768] + b[tid + 768]);
}

// ------------------------------- WMMA GEMM ---------------------------------
// C[M,N] = act(A_bf16[M,K] @ W_f32[K,N] + bias).  Block tile 128x64, 8 waves,
// each wave a 32x32 tile (2 A-frags x 2 B-frags -> 4 WMMA per K-step of 32).
// A staged by async global->LDS B128; fp32 weights converted to bf16 while
// staged (transposed) through VGPRs.
template <bool GELU_ACT, bool RESID, bool OUTB>
__global__ __launch_bounds__(256) void wmma_gemm_kernel(
    const u16* __restrict__ A, const float* __restrict__ Wt,
    const float* __restrict__ bias, u16* __restrict__ outB,
    float* __restrict__ outF, int M, int N, int K) {
  __shared__ u16 lsA[128 * 40];  // [row][k]  stride 40 halves (80B, 16B aligned)
  __shared__ u16 lsB[64 * 40];   // [n][k]    transposed weights

  const int tid  = threadIdx.x, lane = tid & 31, wid = tid >> 5;
  const int wm   = wid >> 1, wn = wid & 1;
  const int row0 = blockIdx.y * 128, col0 = blockIdx.x * 64;

  f32x8 acc00 = (f32x8)(0.0f), acc01 = (f32x8)(0.0f);
  f32x8 acc10 = (f32x8)(0.0f), acc11 = (f32x8)(0.0f);

  const int ar = tid >> 1, ac = (tid & 1) * 16;  // A tile: 128 rows x 32 halves
  const int bk = tid >> 3, bn = (tid & 7) * 8;   // W tile: 32 k    x 64 floats

  const int am0  = wm * 32 + (lane & 15);
  const int am1  = am0 + 16;
  const int c0   = (lane < 16) ? 0 : 8;
  const int base = (lane < 16) ? 0 : 16;
  const int n0   = wn * 32 + (lane & 15);

  for (int k0 = 0; k0 < K; k0 += 32) {
    __syncthreads();
    // stage A via async global->LDS (raw bf16 copy, no VGPR staging)
    const u16* ap = A + (size_t)(row0 + ar) * K + k0 + ac;
    async_ld_b128(&lsA[ar * 40 + ac],     ap);
    async_ld_b128(&lsA[ar * 40 + ac + 8], ap + 8);
    // stage W (fp32 -> bf16, transposed so K is contiguous per output column)
    const float* wp = Wt + (size_t)(k0 + bk) * N + col0 + bn;
    float4 w0 = *(const float4*)(wp);
    float4 w1 = *(const float4*)(wp + 4);
    lsB[(bn + 0) * 40 + bk] = f2bf(w0.x);
    lsB[(bn + 1) * 40 + bk] = f2bf(w0.y);
    lsB[(bn + 2) * 40 + bk] = f2bf(w0.z);
    lsB[(bn + 3) * 40 + bk] = f2bf(w0.w);
    lsB[(bn + 4) * 40 + bk] = f2bf(w1.x);
    lsB[(bn + 5) * 40 + bk] = f2bf(w1.y);
    lsB[(bn + 6) * 40 + bk] = f2bf(w1.z);
    lsB[(bn + 7) * 40 + bk] = f2bf(w1.w);
    if (k0 + 32 < K) {   // global_prefetch_b8 for next tiles
      __builtin_prefetch(ap + 32, 0, 3);
      __builtin_prefetch(Wt + (size_t)(k0 + 32 + bk) * N + col0 + bn, 0, 3);
    }
    wait_async0();       // A-tile async copies complete before the barrier
    __syncthreads();

    // A fragments: halves 0-7 = K c0..c0+7, halves 8-15 = K c0+16..c0+23
    Frag16 a0, a1;
    a0.q[0] = *(const uint4*)&lsA[am0 * 40 + c0];
    a0.q[1] = *(const uint4*)&lsA[am0 * 40 + c0 + 16];
    a1.q[0] = *(const uint4*)&lsA[am1 * 40 + c0];
    a1.q[1] = *(const uint4*)&lsA[am1 * 40 + c0 + 16];
    // B fragments: halves 0-7 = K c0..c0+7, halves 8-15 = K c0+16..c0+23
    Frag16 b0, b1;
    b0.q[0] = *(const uint4*)&lsB[n0 * 40 + c0];
    b0.q[1] = *(const uint4*)&lsB[n0 * 40 + c0 + 16];
    b1.q[0] = *(const uint4*)&lsB[(n0 + 16) * 40 + c0];
    b1.q[1] = *(const uint4*)&lsB[(n0 + 16) * 40 + c0 + 16];

    acc00 = WMMA_BF16(a0.v, b0.v, acc00);
    acc01 = WMMA_BF16(a0.v, b1.v, acc01);
    acc10 = WMMA_BF16(a1.v, b0.v, acc10);
    acc11 = WMMA_BF16(a1.v, b1.v, acc11);
  }

  // epilogue: C layout -> row = r + 8*(lane>=16), col = 16-lane group index
  const int rb0 = row0 + wm * 32 + ((lane >> 4) << 3);
  const int rb1 = rb0 + 16;
  const int cA  = col0 + wn * 32 + (lane & 15);
  const int cB  = cA + 16;
  const float bA = bias[cA], bB = bias[cB];
#pragma unroll
  for (int r = 0; r < 8; ++r) {
    float v00 = acc00[r] + bA, v01 = acc01[r] + bB;
    float v10 = acc10[r] + bA, v11 = acc11[r] + bB;
    if (GELU_ACT) {
      v00 = 0.5f * v00 * (1.0f + erff(v00 * 0.70710678118f));
      v01 = 0.5f * v01 * (1.0f + erff(v01 * 0.70710678118f));
      v10 = 0.5f * v10 * (1.0f + erff(v10 * 0.70710678118f));
      v11 = 0.5f * v11 * (1.0f + erff(v11 * 0.70710678118f));
    }
    if (RESID) {
      outF[(size_t)(rb0 + r) * N + cA] += v00;
      outF[(size_t)(rb0 + r) * N + cB] += v01;
      outF[(size_t)(rb1 + r) * N + cA] += v10;
      outF[(size_t)(rb1 + r) * N + cB] += v11;
    }
    if (OUTB) {
      outB[(size_t)(rb0 + r) * N + cA] = f2bf(v00);
      outB[(size_t)(rb0 + r) * N + cB] = f2bf(v01);
      outB[(size_t)(rb1 + r) * N + cA] = f2bf(v10);
      outB[(size_t)(rb1 + r) * N + cB] = f2bf(v11);
    }
  }
}

// ----------------------------- Flash attention -----------------------------
// QKV bf16 [B*T, 3072], head layout: channel block h*192 (+0 q, +64 k, +128 v).
// Block = 8 waves, each wave owns a 16-row Q tile; all waves share (b,h) so
// K/V tiles are staged once per block per step.  K tile staged by async
// global->LDS; V tile transposed through VGPRs.  Online softmax in fp32;
// scores scaled by SCALE^2 = 1/8.  Row-sums of P computed by WMMA vs ones.
__global__ __launch_bounds__(256) void attn_kernel(
    const u16* __restrict__ QKV, u16* __restrict__ AO) {
  __shared__ u16   lsK[32 * 72];        // [key][ch]   stride 72 halves (144B)
  __shared__ u16   lsV[64 * 40];        // [ch][key]   transposed
  __shared__ float lsP[8 * 16 * 40];    // per-wave P staging, stride 40 floats

  const int tid = threadIdx.x, lane = tid & 31, wq = tid >> 5;
  const int b = blockIdx.y >> 4, h = blockIdx.y & 15;
  const int t0 = blockIdx.x * 128;

  const int m    = lane & 15;
  const int c0   = (lane < 16) ? 0 : 8;
  const int base = (lane < 16) ? 0 : 16;
  const int rl   = (lane >> 4) << 3;

  // Q fragments for this wave's 16 rows (ch 0-31 and 32-63)
  Frag16 qF0, qF1;
  {
    const u16* qp = QKV + (size_t)(b * 1024 + t0 + wq * 16 + m) * 3072 + h * 192;
    qF0.q[0] = *(const uint4*)(qp + c0);
    qF0.q[1] = *(const uint4*)(qp + c0 + 16);
    qF1.q[0] = *(const uint4*)(qp + 32 + c0);
    qF1.q[1] = *(const uint4*)(qp + 32 + c0 + 16);
  }

  // ones matrix (B-frag) for WMMA row-sum of P
  Frag16 ones;
#pragma unroll
  for (int i = 0; i < 16; ++i) ones.s[i] = 0x3F80;   // bf16 1.0

  f32x8 o0 = (f32x8)(0.0f), o1 = (f32x8)(0.0f), o2 = (f32x8)(0.0f), o3 = (f32x8)(0.0f);
  float mrun[8], lrun[8], corr[8];
#pragma unroll
  for (int r = 0; r < 8; ++r) { mrun[r] = -1e30f; lrun[r] = 0.0f; }

  const int ld_key = tid >> 3, ld_c8 = (tid & 7) * 8;
  float* pstage = &lsP[wq * 640];

  for (int j = 0; j < 1024; j += 32) {
    __syncthreads();
    // cooperative K/V staging (32 keys x 64 ch)
    const u16* kp = QKV + (size_t)(b * 1024 + j + ld_key) * 3072 + h * 192;
    async_ld_b128(&lsK[ld_key * 72 + ld_c8], kp + 64 + ld_c8);   // K: raw copy
    uint4 vv = *(const uint4*)(kp + 128 + ld_c8);                // V: transpose
    const u16* vs = (const u16*)&vv;
#pragma unroll
    for (int i = 0; i < 8; ++i) lsV[(ld_c8 + i) * 40 + ld_key] = vs[i];
    wait_async0();
    __syncthreads();

    // S = Q . K^T for 32 keys (two 16-key halves, two 32-ch chunks each)
    f32x8 s0 = (f32x8)(0.0f), s1 = (f32x8)(0.0f);
    Frag16 kf;
    kf.q[0] = *(const uint4*)&lsK[m * 72 + c0];
    kf.q[1] = *(const uint4*)&lsK[m * 72 + c0 + 16];
    s0 = WMMA_BF16(qF0.v, kf.v, s0);
    kf.q[0] = *(const uint4*)&lsK[m * 72 + 32 + c0];
    kf.q[1] = *(const uint4*)&lsK[m * 72 + 32 + c0 + 16];
    s0 = WMMA_BF16(qF1.v, kf.v, s0);
    kf.q[0] = *(const uint4*)&lsK[(m + 16) * 72 + c0];
    kf.q[1] = *(const uint4*)&lsK[(m + 16) * 72 + c0 + 16];
    s1 = WMMA_BF16(qF0.v, kf.v, s1);
    kf.q[0] = *(const uint4*)&lsK[(m + 16) * 72 + 32 + c0];
    kf.q[1] = *(const uint4*)&lsK[(m + 16) * 72 + 32 + c0 + 16];
    s1 = WMMA_BF16(qF1.v, kf.v, s1);

    // online softmax max + rescale (row lives in a 16-lane group; xor masks
    // 1/2/4/8 stay in-group on wave32)
#pragma unroll
    for (int r = 0; r < 8; ++r) {
      float a = s0[r] * 0.125f, c = s1[r] * 0.125f;   // SCALE^2 = 1/sqrt(CH)
      float mx = fmaxf(a, c);
      mx = fmaxf(mx, __shfl_xor(mx, 1));
      mx = fmaxf(mx, __shfl_xor(mx, 2));
      mx = fmaxf(mx, __shfl_xor(mx, 4));
      mx = fmaxf(mx, __shfl_xor(mx, 8));
      const float nm = fmaxf(mrun[r], mx);
      const float p0 = __expf(a - nm), p1 = __expf(c - nm);
      corr[r] = __expf(mrun[r] - nm);
      mrun[r] = nm;
      o0[r] *= corr[r]; o1[r] *= corr[r]; o2[r] *= corr[r]; o3[r] *= corr[r];
      pstage[(r + rl) * 40 + m]      = p0;
      pstage[(r + rl) * 40 + 16 + m] = p1;
    }
    // wave-local LDS store->load hazard: drain DScnt explicitly
    asm volatile("s_wait_dscnt 0x0" ::: "memory");

    // repack P (C-layout in LDS) into an A-matrix fragment, f32 -> bf16
    Frag16 pF;
    {
      const float* pr = &pstage[m * 40];
      float4 f0 = *(const float4*)(pr + c0);
      float4 f1 = *(const float4*)(pr + c0 + 4);
      float4 f2 = *(const float4*)(pr + c0 + 16);
      float4 f3 = *(const float4*)(pr + c0 + 20);
      pF.s[0]  = f2bf(f0.x); pF.s[1]  = f2bf(f0.y); pF.s[2]  = f2bf(f0.z); pF.s[3]  = f2bf(f0.w);
      pF.s[4]  = f2bf(f1.x); pF.s[5]  = f2bf(f1.y); pF.s[6]  = f2bf(f1.z); pF.s[7]  = f2bf(f1.w);
      pF.s[8]  = f2bf(f2.x); pF.s[9]  = f2bf(f2.y); pF.s[10] = f2bf(f2.z); pF.s[11] = f2bf(f2.w);
      pF.s[12] = f2bf(f3.x); pF.s[13] = f2bf(f3.y); pF.s[14] = f2bf(f3.z); pF.s[15] = f2bf(f3.w);
    }

    // row-sums of P via WMMA against ones: result replicated over N-group,
    // landing in exactly the per-row register layout lrun uses.
    f32x8 rsum = (f32x8)(0.0f);
    rsum = WMMA_BF16(pF.v, ones.v, rsum);
#pragma unroll
    for (int r = 0; r < 8; ++r) lrun[r] = lrun[r] * corr[r] + rsum[r];

    // O += P . V  (four 16-channel chunks)
    Frag16 vf;
    vf.q[0] = *(const uint4*)&lsV[m * 40 + c0];
    vf.q[1] = *(const uint4*)&lsV[m * 40 + c0 + 16];
    o0 = WMMA_BF16(pF.v, vf.v, o0);
    vf.q[0] = *(const uint4*)&lsV[(m + 16) * 40 + c0];
    vf.q[1] = *(const uint4*)&lsV[(m + 16) * 40 + c0 + 16];
    o1 = WMMA_BF16(pF.v, vf.v, o1);
    vf.q[0] = *(const uint4*)&lsV[(m + 32) * 40 + c0];
    vf.q[1] = *(const uint4*)&lsV[(m + 32) * 40 + c0 + 16];
    o2 = WMMA_BF16(pF.v, vf.v, o2);
    vf.q[0] = *(const uint4*)&lsV[(m + 48) * 40 + c0];
    vf.q[1] = *(const uint4*)&lsV[(m + 48) * 40 + c0 + 16];
    o3 = WMMA_BF16(pF.v, vf.v, o3);
  }

  // normalize + write bf16 attention output at channel block h*64
#pragma unroll
  for (int r = 0; r < 8; ++r) {
    const int row = t0 + wq * 16 + r + rl;
    const float inv = 1.0f / lrun[r];
    u16* op = AO + (size_t)(b * 1024 + row) * 1024 + h * 64;
    op[m]      = f2bf(o0[r] * inv);
    op[16 + m] = f2bf(o1[r] * inv);
    op[32 + m] = f2bf(o2[r] * inv);
    op[48 + m] = f2bf(o3[r] * inv);
  }
}

// ------------------------------- launcher ----------------------------------
extern "C" void kernel_launch(void* const* d_in, const int* in_sizes, int n_in,
                              void* d_out, int out_size, void* d_ws, size_t ws_size,
                              hipStream_t stream) {
  (void)in_sizes; (void)n_in; (void)out_size; (void)ws_size;
  constexpr int L = 12, W = 1024, T = 1024, Bb = 2, M = Bb * T;
  constexpr size_t HF_BYTES = (size_t)M * W * 4;

  const float* x      = (const float*)d_in[0];
  const float* qkv_w  = (const float*)d_in[1];
  const float* qkv_b  = (const float*)d_in[2];
  const float* proj_w = (const float*)d_in[3];
  const float* proj_b = (const float*)d_in[4];
  const float* ln1_g  = (const float*)d_in[5];
  const float* ln1_b  = (const float*)d_in[6];
  const float* ln2_g  = (const float*)d_in[7];
  const float* ln2_b  = (const float*)d_in[8];
  const float* fc_w   = (const float*)d_in[9];
  const float* fc_b   = (const float*)d_in[10];
  const float* fc2_w  = (const float*)d_in[11];
  const float* fc2_b  = (const float*)d_in[12];

  char* ws = (char*)d_ws;
  float* Hf  = (float*)(ws);                               // 8 MB  fp32 residual
  u16*  Xbf  = (u16*)(ws + 8388608);                       // 4 MB  LN output
  u16*  QKV  = (u16*)(ws + 12582912);                      // 12 MB qkv
  u16*  AO   = (u16*)(ws + 25165824);                      // 4 MB  attn out
  u16*  Z1   = (u16*)(ws + 29360128);                      // 16 MB fc1 out

  hipMemcpyAsync(Hf, x, HF_BYTES, hipMemcpyDeviceToDevice, stream);

  const dim3 blk(256);
  for (int l = 0; l < L; ++l) {
    // h = h + attn(ln1(h))
    ln_bf16_kernel<<<M, blk, 0, stream>>>(Hf, ln1_g + l * W, ln1_b + l * W, Xbf);
    wmma_gemm_kernel<false, false, true><<<dim3(48, 16), blk, 0, stream>>>(
        Xbf, qkv_w + (size_t)l * W * 3 * W, qkv_b + (size_t)l * 3 * W,
        QKV, nullptr, M, 3 * W, W);
    attn_kernel<<<dim3(8, 32), blk, 0, stream>>>(QKV, AO);
    wmma_gemm_kernel<false, true, false><<<dim3(16, 16), blk, 0, stream>>>(
        AO, proj_w + (size_t)l * W * W, proj_b + (size_t)l * W,
        nullptr, Hf, M, W, W);
    // h = h + fc2(gelu(fc1(ln2(h))))
    ln_bf16_kernel<<<M, blk, 0, stream>>>(Hf, ln2_g + l * W, ln2_b + l * W, Xbf);
    wmma_gemm_kernel<true, false, true><<<dim3(64, 16), blk, 0, stream>>>(
        Xbf, fc_w + (size_t)l * W * 4 * W, fc_b + (size_t)l * 4 * W,
        Z1, nullptr, M, 4 * W, W);
    wmma_gemm_kernel<false, true, false><<<dim3(16, 16), blk, 0, stream>>>(
        Z1, fc2_w + (size_t)l * 4 * W * W, fc2_b + (size_t)l * W,
        nullptr, Hf, M, W, 4 * W);
  }

  hipMemcpyAsync(d_out, Hf, HF_BYTES, hipMemcpyDeviceToDevice, stream);
}
